// IN_32615981646573
// MI455X (gfx1250) — compile-verified, matrix operand order checked by reading, not run on hardware
//
#include <hip/hip_runtime.h>
#include <math.h>

// ---------------------------------------------------------------------------
// Fused interaction-network forward for MI455X (gfx1250, wave32, WMMA).
// One workgroup per batch element; 8 waves.
//  * GEMM1 computed swapped (H^T = W1 x A^T) so its D-layout IS the A-fragment
//    layout of GEMM2 -> relu/bias/cvt entirely in registers (no H round-trip).
//  * Gathers go straight into fragment registers: one unconditional
//    ds_load_b32 from a fused x|y|Ebar LDS array + one cndmask per element.
//  * /29 replaced by an LDS recv|send pair table; /870 replaced by a per-tile
//    carry decomposition (r is per-lane constant).
//  * Weight fragments + per-lane bias vectors hoisted into registers per
//    branch; WMMAs issue back-to-back on register operands.
// ---------------------------------------------------------------------------

#define HIDDEN 60
#define NP_    30
#define PDIM   30
#define NV_    14
#define SDIM   14
#define DE_    20
#define DO_    24
#define NPP    870   // 30*29 ordered pairs (i != j)
#define NPV    420   // 30*14
#define TILES_PP 55  // ceil(870/16)
#define TILES_PV 27  // ceil(420/16)
#define TILES_O  2   // ceil(30/16)

// fused LDS data array: x | y | Ebar
#define XOFF 0
#define YOFF 900            // 30*30
#define EOFF 1096           // 900 + 14*14
#define XYTOT 1696          // + 30*20

typedef __attribute__((ext_vector_type(16))) _Float16 v16h;
typedef __attribute__((ext_vector_type(8)))  _Float16 v8h;
typedef __attribute__((ext_vector_type(8)))  float    v8f;

// Load one 16x32 f16 WMMA operand fragment from an LDS matrix stored
// row-major [rows][64 halves]. CDNA5 16-bit operand layout:
//   lane L -> row L%16 ; element e -> K = ks*32 + (e/8)*16 + 8*(L/16) + e%8
__device__ __forceinline__ v16h frag16(const _Float16* base, int lane, int ks) {
  const int row = lane & 15;
  const int grp = (lane >> 4) & 1;
  const _Float16* p = base + row * 64 + ks * 32 + grp * 8;
  v8h lo = *(const v8h*)(p);
  v8h hi = *(const v8h*)(p + 16);
  v16h f;
#pragma unroll
  for (int i = 0; i < 8; ++i) { f[i] = lo[i]; f[i + 8] = hi[i]; }
  return f;
}

__device__ __forceinline__ v8f wmma16(v16h a, v16h b, v8f c) {
  return __builtin_amdgcn_wmma_f32_16x16x32_f16(false, a, false, b,
                                                (short)0, c, false, false);
}

__device__ __forceinline__ void load_layer(const _Float16* W1, const float* B1,
                                           const _Float16* W2, int lane,
                                           v16h w1f[2][4], v8f b1f[4],
                                           v16h w2f[2][2]) {
  const int g = lane >> 4;
#pragma unroll
  for (int ks = 0; ks < 2; ++ks) {
#pragma unroll
    for (int nt = 0; nt < 4; ++nt) w1f[ks][nt] = frag16(W1 + nt * 1024, lane, ks);
#pragma unroll
    for (int ct = 0; ct < 2; ++ct) w2f[ks][ct] = frag16(W2 + ct * 1024, lane, ks);
  }
#pragma unroll
  for (int nt = 0; nt < 4; ++nt)
#pragma unroll
    for (int e = 0; e < 8; ++e) b1f[nt][e] = B1[nt * 16 + 8 * g + e];
}

// Two-layer MLP on one 16-row tile, all operands in registers.
__device__ __forceinline__ void two_layer(const v16h af[2],
                                          const v16h w1f[2][4],
                                          const v8f  b1f[4],
                                          const v16h w2f[2][2],
                                          v8f& e0, v8f& e1) {
  v8f acc[4] = {};
#pragma unroll
  for (int ks = 0; ks < 2; ++ks)
#pragma unroll
    for (int nt = 0; nt < 4; ++nt)
      acc[nt] = wmma16(w1f[ks][nt], af[ks], acc[nt]);
#pragma unroll
  for (int ks = 0; ks < 2; ++ks) {
    v16h hf;
#pragma unroll
    for (int e = 0; e < 8; ++e) {
      hf[e]     = (_Float16)fmaxf(acc[2 * ks][e]     + b1f[2 * ks][e],     0.f);
      hf[e + 8] = (_Float16)fmaxf(acc[2 * ks + 1][e] + b1f[2 * ks + 1][e], 0.f);
    }
    e0 = wmma16(hf, w2f[ks][0], e0);
    e1 = wmma16(hf, w2f[ks][1], e1);
  }
}

// K index for fragment element e of K-step ks (g = lane/16).
__device__ __forceinline__ int kidx_of(int ks, int g, int e) {
  return ks * 32 + ((e & 8) << 1) + 8 * g + (e & 7);
}

__global__ __launch_bounds__(256, 1)
void interaction_net_kernel(const float* __restrict__ gx,   const float* __restrict__ gy,
                            const float* __restrict__ w1pp, const float* __restrict__ b1pp,
                            const float* __restrict__ w2pp, const float* __restrict__ b2pp,
                            const float* __restrict__ w1pv, const float* __restrict__ b1pv,
                            const float* __restrict__ w2pv, const float* __restrict__ b2pv,
                            const float* __restrict__ w1o,  const float* __restrict__ b1o,
                            const float* __restrict__ w2o,  const float* __restrict__ b2o,
                            const float* __restrict__ wc,   const float* __restrict__ bc,
                            float* __restrict__ out) {
  __shared__ __align__(16) float    s_xy[XYTOT];         // x | y | Ebar
  __shared__ unsigned               s_pair[NPP];         // recv | send<<16
  __shared__ __align__(16) _Float16 s_w1pp[64 * 64];     // [n][k], f16, padded
  __shared__ __align__(16) _Float16 s_w2pp[32 * 64];
  __shared__ __align__(16) _Float16 s_w1pv[64 * 64];
  __shared__ __align__(16) _Float16 s_w2pv[32 * 64];
  __shared__ __align__(16) _Float16 s_w1o [64 * 64];
  __shared__ __align__(16) _Float16 s_w2o [32 * 64];
  __shared__ float s_b1pp[64], s_b2pp[32];
  __shared__ float s_b1pv[64], s_b2pv[32];
  __shared__ float s_b1o [64], s_b2o [32];
  __shared__ float s_wc[DO_];
  __shared__ float s_Osum[DO_];

  const int b    = blockIdx.x;
  const int tid  = threadIdx.x;
  const int lane = tid & 31;
  const int wave = tid >> 5;
  const int n0   = lane & 15;
  const int g    = lane >> 4;

  // ------------------- stage inputs + weights into LDS -------------------
  {
    const float* xb = gx + (size_t)b * PDIM * NP_;
    for (int e = tid; e < PDIM * NP_; e += 256) s_xy[XOFF + e] = xb[e];
    const float* yb = gy + (size_t)b * SDIM * NV_;
    for (int e = tid; e < SDIM * NV_; e += 256) s_xy[YOFF + e] = yb[e];
    for (int e = tid; e < NP_ * DE_; e += 256)  s_xy[EOFF + e] = 0.f;

    for (int q = tid; q < NPP; q += 256) {
      const int i0 = q / 29;
      const int kk = q - i0 * 29;
      const int j  = kk + (kk >= i0);
      s_pair[q] = (unsigned)i0 | ((unsigned)j << 16);
    }

    for (int e = tid; e < 64 * 64; e += 256) {
      int n = e >> 6, k = e & 63;
      s_w1pp[e] = (_Float16)((n < 60 && k < 60) ? w1pp[n * 60 + k] : 0.f);
      s_w1pv[e] = (_Float16)((n < 60 && k < 44) ? w1pv[n * 44 + k] : 0.f);
      s_w1o [e] = (_Float16)((n < 60 && k < 50) ? w1o [n * 50 + k] : 0.f);
    }
    for (int e = tid; e < 32 * 64; e += 256) {
      int n = e >> 6, k = e & 63;
      s_w2pp[e] = (_Float16)((n < 20 && k < 60) ? w2pp[n * 60 + k] : 0.f);
      s_w2pv[e] = (_Float16)((n < 20 && k < 60) ? w2pv[n * 60 + k] : 0.f);
      s_w2o [e] = (_Float16)((n < 24 && k < 60) ? w2o [n * 60 + k] : 0.f);
    }
    if (tid < 64) {
      s_b1pp[tid] = (tid < 60) ? b1pp[tid] : 0.f;
      s_b1pv[tid] = (tid < 60) ? b1pv[tid] : 0.f;
      s_b1o [tid] = (tid < 60) ? b1o [tid] : 0.f;
    }
    if (tid < 32) {
      s_b2pp[tid] = (tid < 20) ? b2pp[tid] : 0.f;
      s_b2pv[tid] = (tid < 20) ? b2pv[tid] : 0.f;
      s_b2o [tid] = (tid < 24) ? b2o [tid] : 0.f;
    }
    if (tid < DO_) { s_wc[tid] = wc[tid]; s_Osum[tid] = 0.f; }
  }
  __syncthreads();

  float* s_Ebar = s_xy + EOFF;

  // ------------------------------ pp branch ------------------------------
  // Row r covers flat indices m = r*60 + c of the (60, 870) Bm array:
  // (f, q) = divmod(m, 870); value = f<30 ? x[f, recv[q]] : x[f-30, send[q]].
  {
    v16h w1f[2][4], w2f[2][2]; v8f b1f[4];
    load_layer(s_w1pp, s_b1pp, s_w2pp, lane, w1f, b1f, w2f);
    const float bi0 = s_b2pp[n0];                      // n0 < 16 < DE
    const float bi1 = (n0 < DE_ - 16) ? s_b2pp[16 + n0] : 0.f;

    for (int t = wave; t < TILES_PP; t += 8) {
      const int r  = t * 16 + n0;                      // per-lane row
      const int rb = r * 60;
      const int a  = rb / NPP;                         // once per tile
      const int b0 = rb - a * NPP;
      const bool rok = (r < NPP);
      v16h af[2];
#pragma unroll
      for (int ks = 0; ks < 2; ++ks) {
#pragma unroll
        for (int e = 0; e < 16; ++e) {
          const int c  = kidx_of(ks, g, e);
          const int t1 = b0 + c;                       // < 870 + 64
          const bool cy = (t1 >= NPP);
          const int f  = a + (cy ? 1 : 0);
          const int q  = t1 - (cy ? NPP : 0);          // always < 870
          const unsigned tb = s_pair[q];
          const bool lo = (f < 30);
          const int idx  = lo ? (int)(tb & 0xffffu) : (int)(tb >> 16);
          const int feat = lo ? f : f - 30;
          const float raw = s_xy[feat * NP_ + idx];    // unconditional ds load
          af[ks][e] = (_Float16)((rok && c < 60) ? raw : 0.f);
        }
      }

      v8f e0 = {}, e1 = {};
      two_layer(af, w1f, b1f, w2f, e0, e1);

#pragma unroll
      for (int v = 0; v < 8; ++v) {
        const int p = t * 16 + v + 8 * g;
        if (p < NPP) {
          float* dst = s_Ebar + (p / 29) * DE_;
          atomicAdd(dst + n0, e0[v] + bi0);
          if (n0 < DE_ - 16) atomicAdd(dst + 16 + n0, e1[v] + bi1);
        }
      }
    }
  }

  // ------------------------------ pv branch ------------------------------
  // Row p, col c: c<30 -> x[c, p/14] ; c<44 -> y[c-30, p%14]
  {
    v16h w1f[2][4], w2f[2][2]; v8f b1f[4];
    load_layer(s_w1pv, s_b1pv, s_w2pv, lane, w1f, b1f, w2f);
    const float bi0 = s_b2pv[n0];
    const float bi1 = (n0 < DE_ - 16) ? s_b2pv[16 + n0] : 0.f;

    for (int t = wave; t < TILES_PV; t += 8) {
      const int p  = t * 16 + n0;
      const int kq = p / NV_;
      const int vq = p - kq * NV_;
      const bool pok = (p < NPV);
      v16h af[2];
#pragma unroll
      for (int ks = 0; ks < 2; ++ks) {
#pragma unroll
        for (int e = 0; e < 16; ++e) {
          const int c = kidx_of(ks, g, e);
          const bool cx = (c < 30);
          const int addr = cx ? (c * NP_ + kq)
                              : (YOFF + (c - 30) * NV_ + vq);
          const float raw = s_xy[addr];                // unconditional ds load
          af[ks][e] = (_Float16)((pok && c < 44) ? raw : 0.f);
        }
      }

      v8f e0 = {}, e1 = {};
      two_layer(af, w1f, b1f, w2f, e0, e1);

#pragma unroll
      for (int v = 0; v < 8; ++v) {
        const int p2 = t * 16 + v + 8 * g;
        if (p2 < NPV) {
          float* dst = s_Ebar + (p2 / NV_) * DE_;
          atomicAdd(dst + n0, e0[v] + bi0);
          if (n0 < DE_ - 16) atomicAdd(dst + 16 + n0, e1[v] + bi1);
        }
      }
    }
  }

  __syncthreads();  // Ebar complete

  // ------------------------------- o branch ------------------------------
  // J row i: [x[i, 0..29] , Ebar[i, 0..19]]; outputs row-summed into Osum.
  {
    v16h w1f[2][4], w2f[2][2]; v8f b1f[4];
    load_layer(s_w1o, s_b1o, s_w2o, lane, w1f, b1f, w2f);
    const float bi0 = s_b2o[n0];
    const float bi1 = (n0 < DO_ - 16) ? s_b2o[16 + n0] : 0.f;

    for (int t = wave; t < TILES_O; t += 8) {
      const int r = t * 16 + n0;
      const bool rok = (r < NP_);
      v16h af[2];
#pragma unroll
      for (int ks = 0; ks < 2; ++ks) {
#pragma unroll
        for (int e = 0; e < 16; ++e) {
          const int c = kidx_of(ks, g, e);
          const bool cx = (c < 30);
          int addr = cx ? (r * NP_ + c) : (EOFF + r * DE_ + (c - 30));
          addr = (addr < XYTOT) ? addr : 0;            // clamp (r pad rows)
          const float raw = s_xy[addr];
          af[ks][e] = (_Float16)((rok && c < 50) ? raw : 0.f);
        }
      }

      v8f e0 = {}, e1 = {};
      two_layer(af, w1f, b1f, w2f, e0, e1);

#pragma unroll
      for (int v = 0; v < 8; ++v) {
        const int p = t * 16 + v + 8 * g;
        if (p < NP_) {
          atomicAdd(&s_Osum[n0], e0[v] + bi0);
          if (n0 < DO_ - 16) atomicAdd(&s_Osum[16 + n0], e1[v] + bi1);
        }
      }
    }
  }

  __syncthreads();

  if (tid == 0) {
    float s = bc[0];
#pragma unroll
    for (int n = 0; n < DO_; ++n) s += s_Osum[n] * s_wc[n];
    out[b] = 1.f / (1.f + __expf(-s));
  }
}

extern "C" void kernel_launch(void* const* d_in, const int* in_sizes, int n_in,
                              void* d_out, int out_size, void* d_ws, size_t ws_size,
                              hipStream_t stream) {
  (void)n_in; (void)d_ws; (void)ws_size; (void)out_size;
  const int batch = in_sizes[0] / (PDIM * NP_);   // 1024
  interaction_net_kernel<<<batch, 256, 0, stream>>>(
      (const float*)d_in[0],  (const float*)d_in[1],
      (const float*)d_in[2],  (const float*)d_in[3],
      (const float*)d_in[4],  (const float*)d_in[5],
      (const float*)d_in[6],  (const float*)d_in[7],
      (const float*)d_in[8],  (const float*)d_in[9],
      (const float*)d_in[10], (const float*)d_in[11],
      (const float*)d_in[12], (const float*)d_in[13],
      (const float*)d_in[14], (const float*)d_in[15],
      (float*)d_out);
}
